// LIFLayer_6622839570900
// MI455X (gfx1250) — matrix operand order repaired and checked by reading, not compiled, not run on hardware
//
#include <hip/hip_runtime.h>

// LIF layer fused GEMM + recurrence on gfx1250 WMMA.
// Phase 1 (workspace path): split x and W into bf16 hi/lo planes ONCE.
// Phase 2: each wave owns a 16x16 (batch x out) tile; mem lives in the WMMA
// f32 C/D register layout across the whole T loop; GEMM is bf16x3
// (xh*wh + xl*wh + xh*wl) on v_wmma_f32_16x16x32_bf16, run as TWO independent
// accumulator chains so dependent WMMAs can overlap in the matrix pipe.

#define T_STEPS 64
#define B_DIM   128
#define IN_DIM  2048
#define OUT_DIM 2048
#define DECAY   0.6065306597126334f   // exp(-1/2)
#define THRESH  1.0f

typedef __attribute__((ext_vector_type(16))) __bf16 v16bf;
typedef __attribute__((ext_vector_type(8)))  __bf16 v8bf;
typedef __attribute__((ext_vector_type(8)))  float  v8f;
typedef __attribute__((ext_vector_type(4)))  float  f32x4;

// ---------------------------------------------------------------------------
// Phase 1: elementwise split f32 -> bf16 hi + bf16 lo residual (8 elems/thread)
// ---------------------------------------------------------------------------
__global__ __launch_bounds__(256) void split_bf16_kernel(
    const float* __restrict__ in, __bf16* __restrict__ hi,
    __bf16* __restrict__ lo, unsigned n8)  // n8 = n/8
{
    unsigned i = blockIdx.x * blockDim.x + threadIdx.x;
    if (i >= n8) return;
    size_t base = (size_t)i * 8;
    f32x4 v0 = *(const f32x4*)(in + base);
    f32x4 v1 = *(const f32x4*)(in + base + 4);
    float v[8] = { v0.x, v0.y, v0.z, v0.w, v1.x, v1.y, v1.z, v1.w };
    v8bf h, l;
    #pragma unroll
    for (int e = 0; e < 8; ++e) {
        __bf16 hv = (__bf16)v[e];
        h[e] = hv;
        l[e] = (__bf16)(v[e] - (float)hv);
    }
    *(v8bf*)(hi + base) = h;
    *(v8bf*)(lo + base) = l;
}

// ---------------------------------------------------------------------------
// Phase 2: fused GEMM + LIF scan, loading pre-split bf16 planes directly.
// ---------------------------------------------------------------------------
__global__ __launch_bounds__(128) void lif_fused_wmma_bf16_kernel(
    const __bf16* __restrict__ xhi,  // (T, B, IN)
    const __bf16* __restrict__ xlo,
    const __bf16* __restrict__ whi,  // (OUT, IN)
    const __bf16* __restrict__ wlo,
    const float*  __restrict__ bias, // (OUT)
    float*        __restrict__ out)  // (T, B, OUT)
{
    const int lane   = threadIdx.x & 31;
    const int wave   = threadIdx.x >> 5;      // 4 waves per block
    const int waveM  = wave >> 1;             // 0..1 (batch tiles)
    const int waveN  = wave & 1;              // 0..1 (out tiles)

    const int bbase  = blockIdx.y * 32 + waveM * 16;
    const int obase  = blockIdx.x * 32 + waveN * 16;

    const int laneLo = lane & 15;
    const int laneHi = lane >> 4;             // 0/1: K-half select

    const int brow = bbase + laneLo;          // A-matrix row (batch)
    const int ocol = obase + laneLo;          // B/C-matrix col (out channel)

    // B fragment: W[ocol][k + laneHi*8 + {0..7, 16..23}]
    const __bf16* wh_r = whi + (size_t)ocol * IN_DIM + (laneHi << 3);
    const __bf16* wl_r = wlo + (size_t)ocol * IN_DIM + (laneHi << 3);
    // A fragment: x[t][brow][k + laneHi*8 + {0..7, 16..23}]
    const size_t xoff = (size_t)brow * IN_DIM + (laneHi << 3);

    const float bias_v = bias[ocol];

    v8f mem;
    #pragma unroll
    for (int e = 0; e < 8; ++e) mem[e] = 0.0f;

    for (int t = 0; t < T_STEPS; ++t) {
        const __bf16* xh_r = xhi + ((size_t)t * B_DIM) * IN_DIM + xoff;
        const __bf16* xl_r = xlo + ((size_t)t * B_DIM) * IN_DIM + xoff;

        // Two independent accumulator chains -> overlapping WMMA streams.
        v8f acc_h, acc_l;
        #pragma unroll
        for (int e = 0; e < 8; ++e) { acc_h[e] = bias_v; acc_l[e] = 0.0f; }

        for (int k = 0; k < IN_DIM; k += 32) {
            // A fragments: two 16B runs per plane, merged register-adjacent
            v8bf ah0 = *(const v8bf*)(xh_r + k);
            v8bf ah1 = *(const v8bf*)(xh_r + k + 16);
            v8bf al0 = *(const v8bf*)(xl_r + k);
            v8bf al1 = *(const v8bf*)(xl_r + k + 16);
            // B fragments: two 16B runs per plane, merged register-adjacent
            v8bf bh0 = *(const v8bf*)(wh_r + k);
            v8bf bh1 = *(const v8bf*)(wh_r + k + 16);
            v8bf bl0 = *(const v8bf*)(wl_r + k);
            v8bf bl1 = *(const v8bf*)(wl_r + k + 16);

            if (k + 64 < IN_DIM) {  // WGP-scope prefetch of the next chunk
                __builtin_prefetch(xh_r + k + 64, 0, 3);
                __builtin_prefetch(xl_r + k + 64, 0, 3);
                __builtin_prefetch(wh_r + k + 64, 0, 3);
                __builtin_prefetch(wl_r + k + 64, 0, 3);
            }

            v16bf ah = __builtin_shufflevector(ah0, ah1, 0,1,2,3,4,5,6,7,
                                               8,9,10,11,12,13,14,15);
            v16bf al = __builtin_shufflevector(al0, al1, 0,1,2,3,4,5,6,7,
                                               8,9,10,11,12,13,14,15);
            v16bf bh = __builtin_shufflevector(bh0, bh1, 0,1,2,3,4,5,6,7,
                                               8,9,10,11,12,13,14,15);
            v16bf bl = __builtin_shufflevector(bl0, bl1, 0,1,2,3,4,5,6,7,
                                               8,9,10,11,12,13,14,15);

            // Chain 1: main product.  Chain 2: the two bf16x3 correction terms.
            acc_h = __builtin_amdgcn_wmma_f32_16x16x32_bf16(
                        false, ah, false, bh, (short)0, acc_h, false, false);
            acc_l = __builtin_amdgcn_wmma_f32_16x16x32_bf16(
                        false, al, false, bh, (short)0, acc_l, false, false);
            acc_l = __builtin_amdgcn_wmma_f32_16x16x32_bf16(
                        false, ah, false, bl, (short)0, acc_l, false, false);
        }

        float* out_t = out + ((size_t)t * B_DIM) * OUT_DIM;
        #pragma unroll
        for (int e = 0; e < 8; ++e) {
            float m = mem[e] * DECAY + (acc_h[e] + acc_l[e]);
            float s = (m > THRESH) ? 1.0f : 0.0f;
            m -= s * THRESH;
            mem[e] = m;
            const int row = bbase + e + (laneHi << 3);
            out_t[(size_t)row * OUT_DIM + ocol] = s;
        }
    }
}

// ---------------------------------------------------------------------------
// Fallback (small ws): convert-on-the-fly version.
// ---------------------------------------------------------------------------
__global__ __launch_bounds__(128) void lif_fused_wmma_f32_kernel(
    const float* __restrict__ x, const float* __restrict__ W,
    const float* __restrict__ bias, float* __restrict__ out)
{
    const int lane   = threadIdx.x & 31;
    const int wave   = threadIdx.x >> 5;
    const int waveM  = wave >> 1;
    const int waveN  = wave & 1;
    const int bbase  = blockIdx.y * 32 + waveM * 16;
    const int obase  = blockIdx.x * 32 + waveN * 16;
    const int laneLo = lane & 15;
    const int laneHi = lane >> 4;
    const int brow = bbase + laneLo;
    const int ocol = obase + laneLo;

    const float* wrow = W + (size_t)ocol * IN_DIM + (laneHi << 3);
    const size_t xrow_off = (size_t)brow * IN_DIM + (laneHi << 3);
    const float bias_v = bias[ocol];

    v8f mem;
    #pragma unroll
    for (int e = 0; e < 8; ++e) mem[e] = 0.0f;

    for (int t = 0; t < T_STEPS; ++t) {
        const float* xr = x + ((size_t)t * B_DIM) * IN_DIM + xrow_off;
        v8f acc_h, acc_l;
        #pragma unroll
        for (int e = 0; e < 8; ++e) { acc_h[e] = bias_v; acc_l[e] = 0.0f; }

        for (int k = 0; k < IN_DIM; k += 32) {
            f32x4 a0 = *(const f32x4*)(xr + k);
            f32x4 a1 = *(const f32x4*)(xr + k + 4);
            f32x4 a2 = *(const f32x4*)(xr + k + 16);
            f32x4 a3 = *(const f32x4*)(xr + k + 20);
            const float* wp = wrow + k;
            f32x4 b0 = *(const f32x4*)(wp);
            f32x4 b1 = *(const f32x4*)(wp + 4);
            f32x4 b2 = *(const f32x4*)(wp + 16);
            f32x4 b3 = *(const f32x4*)(wp + 20);

            float af[16] = { a0.x,a0.y,a0.z,a0.w, a1.x,a1.y,a1.z,a1.w,
                             a2.x,a2.y,a2.z,a2.w, a3.x,a3.y,a3.z,a3.w };
            float bf[16] = { b0.x,b0.y,b0.z,b0.w, b1.x,b1.y,b1.z,b1.w,
                             b2.x,b2.y,b2.z,b2.w, b3.x,b3.y,b3.z,b3.w };

            v16bf ah, al, bh, bl;
            #pragma unroll
            for (int i = 0; i < 16; ++i) {
                __bf16 xh = (__bf16)af[i];
                ah[i] = xh;  al[i] = (__bf16)(af[i] - (float)xh);
                __bf16 wh = (__bf16)bf[i];
                bh[i] = wh;  bl[i] = (__bf16)(bf[i] - (float)wh);
            }
            acc_h = __builtin_amdgcn_wmma_f32_16x16x32_bf16(
                        false, ah, false, bh, (short)0, acc_h, false, false);
            acc_l = __builtin_amdgcn_wmma_f32_16x16x32_bf16(
                        false, al, false, bh, (short)0, acc_l, false, false);
            acc_l = __builtin_amdgcn_wmma_f32_16x16x32_bf16(
                        false, ah, false, bl, (short)0, acc_l, false, false);
        }

        float* out_t = out + ((size_t)t * B_DIM) * OUT_DIM;
        #pragma unroll
        for (int e = 0; e < 8; ++e) {
            float m = mem[e] * DECAY + (acc_h[e] + acc_l[e]);
            float s = (m > THRESH) ? 1.0f : 0.0f;
            m -= s * THRESH;
            mem[e] = m;
            const int row = bbase + e + (laneHi << 3);
            out_t[(size_t)row * OUT_DIM + ocol] = s;
        }
    }
}

extern "C" void kernel_launch(void* const* d_in, const int* in_sizes, int n_in,
                              void* d_out, int out_size, void* d_ws, size_t ws_size,
                              hipStream_t stream) {
    const float* x    = (const float*)d_in[0];   // (T, B, IN)
    const float* W    = (const float*)d_in[1];   // (OUT, IN)
    const float* bias = (const float*)d_in[2];   // (OUT)
    float*       out  = (float*)d_out;

    const size_t nW = (size_t)OUT_DIM * IN_DIM;           // 4Mi elems
    const size_t nX = (size_t)T_STEPS * B_DIM * IN_DIM;   // 16Mi elems
    const size_t need = (nW + nX) * 2 * sizeof(__bf16);   // hi+lo planes: 80 MB

    dim3 grid(OUT_DIM / 32, B_DIM / 32);   // 64 x 4 workgroups, 4 waves each

    if (ws_size >= need) {
        __bf16* whi = (__bf16*)d_ws;
        __bf16* wlo = whi + nW;
        __bf16* xhi = wlo + nW;
        __bf16* xlo = xhi + nX;

        split_bf16_kernel<<<(unsigned)((nW / 8 + 255) / 256), 256, 0, stream>>>(
            W, whi, wlo, (unsigned)(nW / 8));
        split_bf16_kernel<<<(unsigned)((nX / 8 + 255) / 256), 256, 0, stream>>>(
            x, xhi, xlo, (unsigned)(nX / 8));

        lif_fused_wmma_bf16_kernel<<<grid, 128, 0, stream>>>(
            xhi, xlo, whi, wlo, bias, out);
    } else {
        lif_fused_wmma_f32_kernel<<<grid, 128, 0, stream>>>(x, W, bias, out);
    }
}
